// GCN_prompt_45397804319434
// MI455X (gfx1250) — matrix operand order repaired and checked by reading, not compiled, not run on hardware
//
#include <hip/hip_runtime.h>
#include <hip/hip_bf16.h>

#define NFEAT 500
#define NHID 128
#define NCLASS 16
#define DEGMAX 64

typedef __bf16 v16bf __attribute__((ext_vector_type(16)));
typedef float  v8f   __attribute__((ext_vector_type(8)));

__device__ __forceinline__ unsigned short f32_to_bf16(float f) {
    unsigned int u = __float_as_uint(f);
    u += 0x7FFFu + ((u >> 16) & 1u);   // round-to-nearest-even
    return (unsigned short)(u >> 16);
}

// ---------------------------------------------------------------------------
// Pack a row-major f32 weight [Krows x Ncols] into bf16 WMMA B-fragments.
// Fragment (kt, nt) layout (ISA 7.12.2, 16-bit B, K x N = 32 x 16):
//   lane L holds column N = nt*16 + L%16; elements i hold K = kt*32 + (half*8 + i) for i<8,
//   K = kt*32 + 16 + half*8 + (i-8) for i>=8 (half = L/16).
// Flat: out[((kt*NT + nt)*32 + lane)*16 + i]
// ---------------------------------------------------------------------------
__global__ void pack_b_kernel(const float* __restrict__ W, int Krows, int Ncols,
                              int NT, int total, unsigned short* __restrict__ out) {
    for (int idx = blockIdx.x * blockDim.x + threadIdx.x; idx < total;
         idx += gridDim.x * blockDim.x) {
        int frag = idx >> 9;        // /512
        int r    = idx & 511;
        int lane = r >> 4;
        int i    = r & 15;
        int kt   = frag / NT;
        int nt   = frag % NT;
        int K    = kt * 32 + ((i < 8) ? ((lane >> 4) * 8 + i) : (16 + (lane >> 4) * 8 + (i - 8)));
        int n    = nt * 16 + (lane & 15);
        float v  = (K < Krows && n < Ncols) ? W[(long long)K * Ncols + n] : 0.f;
        out[idx] = f32_to_bf16(v);
    }
}

// ---------------------------------------------------------------------------
// GEMM: out[M x NT*16] = A[M x Kc] (f32, row-major) @ Bpack (+ bias)
// One block = one 16-row M-tile. Block converts the A tile to bf16 A-fragment
// layout in LDS (ISA 7.12.2 16-bit A 16x32 table), waves stripe over N-tiles,
// inner K loop is v_wmma_f32_16x16x32_bf16.
// ---------------------------------------------------------------------------
__global__ void gemm_bf16_kernel(const float* __restrict__ A, int Mtot, int Kc, int KT,
                                 const unsigned short* __restrict__ Bpack, int NT,
                                 const float* __restrict__ bias,
                                 float* __restrict__ out, int ldout) {
    extern __shared__ unsigned short ldsA[];   // KT*512 bf16 values
    const int m0  = blockIdx.x * 16;
    const int tid = threadIdx.x;
    const int nthreads = blockDim.x;

    // Fill A-fragment LDS. 16-bit A 16x32 (MxK): lane = half*16 + M,
    // elems 0..7 -> K = half*8 + i ; elems 8..15 -> K = 16 + half*8 + (i-8)
    const int tot = KT * 512;
    for (int idx = tid; idx < tot; idx += nthreads) {
        int kt   = idx >> 9;
        int r    = idx & 511;
        int lane = r >> 4;
        int i    = r & 15;
        int half = lane >> 4;
        int M    = lane & 15;
        int kl   = (i < 8) ? (half * 8 + i) : (16 + half * 8 + (i - 8));
        int K    = kt * 32 + kl;
        int row  = m0 + M;
        float v  = (K < Kc && row < Mtot) ? A[(long long)row * Kc + K] : 0.f;
        ldsA[idx] = f32_to_bf16(v);
    }
    __syncthreads();

    const int lane   = tid & 31;
    const int wave   = tid >> 5;
    const int nwaves = nthreads >> 5;
    const int half   = lane >> 4;

    for (int nt = wave; nt < NT; nt += nwaves) {
        v8f acc = {};
        for (int kt = 0; kt < KT; ++kt) {
            v16bf a = *reinterpret_cast<const v16bf*>(&ldsA[(kt * 32 + lane) * 16]);
            v16bf b = *reinterpret_cast<const v16bf*>(
                &Bpack[((long long)(kt * NT + nt) * 32 + lane) * 16]);
            acc = __builtin_amdgcn_wmma_f32_16x16x32_bf16(
                false, a, false, b, (short)0, acc, false, false);
        }
        // C/D layout: VGPR r, lanes 0-15 -> M=r, lanes 16-31 -> M=r+8; N=lane&15
        int col  = nt * 16 + (lane & 15);
        float bv = bias ? bias[col] : 0.f;
#pragma unroll
        for (int r = 0; r < 8; ++r) {
            int row = m0 + r + half * 8;
            if (row < Mtot) out[(long long)row * ldout + col] = acc[r] + bv;
        }
    }
}

// ---------------------------------------------------------------------------
// SpMM scatter: Sout[dst[e], :] += w[e] * Hin[src[e], :]   (atomic f32, L2)
// One wave per edge, lanes stripe over features (coalesced gather).
// ---------------------------------------------------------------------------
__global__ void spmm_kernel(const int* __restrict__ src, const int* __restrict__ dst,
                            const float* __restrict__ w, const float* __restrict__ Hin,
                            float* __restrict__ Sout, int E, int F) {
    long long wid = (long long)blockIdx.x * (blockDim.x >> 5) + (threadIdx.x >> 5);
    if (wid >= E) return;
    int lane = threadIdx.x & 31;
    int s = src[wid];
    int d = dst[wid];
    float wv = w[wid];
    const float* hs = Hin + (long long)s * F;
    float* so = Sout + (long long)d * F;
    for (int f = lane; f < F; f += 32)
        unsafeAtomicAdd(&so[f], wv * hs[f]);   // global_atomic_add_f32
}

__global__ void zero_kernel(float* __restrict__ p, long long n) {
    for (long long i = blockIdx.x * (long long)blockDim.x + threadIdx.x; i < n;
         i += (long long)gridDim.x * blockDim.x)
        p[i] = 0.f;
}

// C = relu(C + bias[col])   (in place), F must be power of two
__global__ void relu_bias_kernel(float* __restrict__ C, const float* __restrict__ bias,
                                 long long n, int Fmask) {
    for (long long i = blockIdx.x * (long long)blockDim.x + threadIdx.x; i < n;
         i += (long long)gridDim.x * blockDim.x)
        C[i] = fmaxf(C[i] + bias[i & Fmask], 0.f);
}

// C = relu(C + b3[col]) + OX   (in place)
__global__ void h3_kernel(float* __restrict__ C, const float* __restrict__ b3,
                          const float* __restrict__ OX, long long n, int Fmask) {
    for (long long i = blockIdx.x * (long long)blockDim.x + threadIdx.x; i < n;
         i += (long long)gridDim.x * blockDim.x)
        C[i] = fmaxf(C[i] + b3[i & Fmask], 0.f) + OX[i];
}

// log_softmax over rows of 16 (with bias b2 added first)
__global__ void logsoftmax16_kernel(const float* __restrict__ S, const float* __restrict__ b,
                                    float* __restrict__ out, int N) {
    int row = blockIdx.x * blockDim.x + threadIdx.x;
    if (row >= N) return;
    float v[NCLASS];
    float m = -__builtin_inff();
#pragma unroll
    for (int c = 0; c < NCLASS; ++c) {
        v[c] = S[(long long)row * NCLASS + c] + b[c];
        m = fmaxf(m, v[c]);
    }
    float s = 0.f;
#pragma unroll
    for (int c = 0; c < NCLASS; ++c) s += expf(v[c] - m);
    float lse = m + logf(s);
#pragma unroll
    for (int c = 0; c < NCLASS; ++c) out[(long long)row * NCLASS + c] = v[c] - lse;
}

// result2 = H3 @ Wd + bd  (per-row length-128 dot product)
__global__ void dot_kernel(const float* __restrict__ H3, const float* __restrict__ Wd,
                           const float* __restrict__ bd, float* __restrict__ out, int N) {
    int row = blockIdx.x * blockDim.x + threadIdx.x;
    if (row >= N) return;
    const float* h = H3 + (long long)row * NHID;
    float acc = 0.f;
#pragma unroll 8
    for (int k = 0; k < NHID; ++k) acc += h[k] * Wd[k];
    out[row] = acc + bd[0];
}

extern "C" void kernel_launch(void* const* d_in, const int* in_sizes, int n_in,
                              void* d_out, int out_size, void* d_ws, size_t ws_size,
                              hipStream_t stream) {
    const float* x    = (const float*)d_in[0];
    const int*   esrc = (const int*)d_in[1];
    const int*   edst = (const int*)d_in[2];
    const float* ew   = (const float*)d_in[3];
    const float* W1   = (const float*)d_in[4];
    const float* b1   = (const float*)d_in[5];
    const float* W2   = (const float*)d_in[6];
    const float* b2   = (const float*)d_in[7];
    const float* W3   = (const float*)d_in[8];
    const float* b3   = (const float*)d_in[9];
    const float* Wd   = (const float*)d_in[10];
    const float* bd   = (const float*)d_in[11];
    const float* Wdeg = (const float*)d_in[12];
    const float* bdeg = (const float*)d_in[13];
    const float* Wds  = (const float*)d_in[14];
    const float* bds  = (const float*)d_in[15];

    const int N = in_sizes[0] / NFEAT;   // 100000
    const int E = in_sizes[1];           // 3200000

    float* out1 = (float*)d_out;                      // [N, 16] log_softmax
    float* out2 = out1 + (size_t)N * NCLASS;          // [N]
    float* out3 = out2 + (size_t)N;                   // [N, 64]

    // ---- workspace carve-up -------------------------------------------------
    char* ws = (char*)d_ws;
    size_t off = 0;
    auto carve = [&](size_t bytes) -> void* {
        void* p = ws + off;
        off += (bytes + 255) & ~(size_t)255;
        return p;
    };
    float* bufA = (float*)carve((size_t)N * NHID * 4);    // XW1, later HW3
    float* bufB = (float*)carve((size_t)N * NHID * 4);    // OX = x@Wds + bds
    float* bufC = (float*)carve((size_t)N * NHID * 4);    // S1 -> H -> S3 -> H3
    float* hw2  = (float*)carve((size_t)N * NCLASS * 4);  // h @ W2
    float* s2   = (float*)carve((size_t)N * NCLASS * 4);  // spmm(hw2)
    unsigned short* pw1   = (unsigned short*)carve(512 * 128 * 2);
    unsigned short* pwds  = (unsigned short*)carve(512 * 128 * 2);
    unsigned short* pw3   = (unsigned short*)carve(128 * 128 * 2);
    unsigned short* pw2   = (unsigned short*)carve(128 * 16 * 2);
    unsigned short* pwdeg = (unsigned short*)carve(128 * 64 * 2);

    // ---- 0) pack weights into bf16 WMMA B-fragment layout -------------------
    auto packLaunch = [&](const float* W, int Krows, int Ncols, int KT, int NT,
                          unsigned short* dst) {
        int total = KT * NT * 512;
        pack_b_kernel<<<(total + 255) / 256, 256, 0, stream>>>(W, Krows, Ncols, NT, total, dst);
    };
    packLaunch(W1,   NFEAT, NHID,   16, 8, pw1);    // K padded 500 -> 512
    packLaunch(Wds,  NFEAT, NHID,   16, 8, pwds);
    packLaunch(W3,   NHID,  NHID,    4, 8, pw3);
    packLaunch(W2,   NHID,  NCLASS,  4, 1, pw2);
    packLaunch(Wdeg, NHID,  DEGMAX,  4, 4, pwdeg);

    const int mblocks = (N + 15) / 16;   // 6250
    auto gemm = [&](const float* A, int Kc, int KT, const unsigned short* Bp, int NT,
                    const float* bias, float* out, int ldout, int block) {
        gemm_bf16_kernel<<<mblocks, block, (size_t)KT * 512 * sizeof(unsigned short), stream>>>(
            A, N, Kc, KT, Bp, NT, bias, out, ldout);
    };

    const long long nh = (long long)N * NHID;
    const int ewb = 1024;   // blocks for elementwise grid-stride kernels
    const int spmm_blocks = (E + 7) / 8;   // 8 waves (edges) per 256-thread block

    // ---- 1) input GEMMs: XW1 = x@W1 ; OX = x@Wds + bds ----------------------
    gemm(x, NFEAT, 16, pw1,  8, nullptr, bufA, NHID, 128);
    gemm(x, NFEAT, 16, pwds, 8, bds,     bufB, NHID, 128);

    // ---- 2) S1 = spmm(XW1) ; H = relu(S1 + b1) ------------------------------
    zero_kernel<<<ewb, 256, 0, stream>>>(bufC, nh);
    spmm_kernel<<<spmm_blocks, 256, 0, stream>>>(esrc, edst, ew, bufA, bufC, E, NHID);
    relu_bias_kernel<<<ewb, 256, 0, stream>>>(bufC, b1, nh, NHID - 1);

    // ---- 3) HW3 = H@W3 ; HW2 = H@W2 -----------------------------------------
    gemm(bufC, NHID, 4, pw3, 8, nullptr, bufA, NHID, 128);
    gemm(bufC, NHID, 4, pw2, 1, nullptr, hw2, NCLASS, 32);

    // ---- 4) S3 = spmm(HW3) ; S2 = spmm(HW2) ---------------------------------
    zero_kernel<<<ewb, 256, 0, stream>>>(bufC, nh);
    zero_kernel<<<ewb, 256, 0, stream>>>(s2, (long long)N * NCLASS);
    spmm_kernel<<<spmm_blocks, 256, 0, stream>>>(esrc, edst, ew, bufA, bufC, E, NHID);
    spmm_kernel<<<spmm_blocks, 256, 0, stream>>>(esrc, edst, ew, hw2, s2, E, NCLASS);

    // ---- 5) H3 = relu(S3 + b3) + OX  (in place in bufC) ---------------------
    h3_kernel<<<ewb, 256, 0, stream>>>(bufC, b3, bufB, nh, NHID - 1);

    // ---- 6) outputs ---------------------------------------------------------
    logsoftmax16_kernel<<<(N + 255) / 256, 256, 0, stream>>>(s2, b2, out1, N);
    dot_kernel<<<(N + 255) / 256, 256, 0, stream>>>(bufC, Wd, bd, out2, N);
    gemm(bufC, NHID, 4, pwdeg, 4, bdeg, out3, DEGMAX, 128);
}
